// LinearAttention_37812892074185
// MI455X (gfx1250) — matrix operand order repaired and checked by reading, not compiled, not run on hardware
//
#include <hip/hip_runtime.h>
#include <hip/hip_bf16.h>

// ---------------------------------------------------------------------------
// Linear attention on MI455X (gfx1250), WMMA-bf16 everywhere.
//  K1: qkv = x @ Wqkv^T (+bias) ; fuse elu+1 ; write
//        q -> pre-swizzled WMMA-B fragment layout (for K3)
//        k -> transposed (D,N) row-major       (A matrix of K2)
//        v -> pre-swizzled WMMA-B fragment layout (for K2)
//  K2: per (b,h): kv^T = [k^T @ v]^T, k_sum fused via ds_add_f32;
//      writes 80x64 augmented A matrix [kv^T ; k_sum ; 0-pad] for K3.
//  K3: out^T = kvT_aug @ q^T ; normalizer = row 64 of the product (free);
//      divide + write y (B,N,C) with 16B stores. No LDS at all.
//  K4: final = y @ Wproj^T (+bias) -> f32 d_out via LDS transpose (b128 out).
// Inner loops: register double-buffered global->LDS staging + grouped
// fragment loads so ds_load_b128 batches ahead of the wmma issue, plus
// global_prefetch one tile ahead.
// ---------------------------------------------------------------------------

typedef __bf16 bf16_t;
typedef __attribute__((ext_vector_type(16))) __bf16 v16bf;
typedef __attribute__((ext_vector_type(8)))  __bf16 v8bf;
typedef __attribute__((ext_vector_type(8)))  float  v8f;

#define DIM    768
#define NHEAD  12
#define HDIM   64
#define BATCH  8
#define SEQ    4096
#define ROWS   (BATCH * SEQ)   // 32768
#define QKV_N  (3 * DIM)       // 2304
#define NBH    (BATCH * NHEAD) // 96
#define KVROWS 80              // 64 kv^T rows + 1 ksum row + 15 zero pad

__device__ __forceinline__ float elu1(float x) {
  return x > 0.0f ? x + 1.0f : __expf(x);
}

__device__ __forceinline__ v8f wmma_bf16(v16bf a, v16bf b, v8f c) {
  return __builtin_amdgcn_wmma_f32_16x16x32_bf16(false, a, false, b,
                                                 (short)0, c, false, false);
}

// A fragment (16x32 MxK bf16) from a row-major tile (LDS or global).
// Lane L<16: row L,   K = {0..15};  L>=16: row L-16, K = {16..31}
// -> two contiguous 16B chunks per lane (2x b128 loads).
__device__ __forceinline__ v16bf frag_a(const bf16_t* t, int row0, int col0,
                                        int ld, int lane) {
  const int m  = lane & 15;
  const int kh = (lane < 16) ? 0 : 16;
  const bf16_t* r = t + (row0 + m) * ld + col0 + kh;
  v16bf a;
#pragma unroll
  for (int e = 0; e < 8; ++e)  a[e] = r[e];        // K = kh + e
#pragma unroll
  for (int e = 8; e < 16; ++e) a[e] = r[e];        // K = kh + e
  return a;
}

// Pre-swizzled B fragment (32x16 KxN bf16): one lane's 16 values contiguous.
//   offset((k,n) in 32xW tile) = (((n>>4)*32) + ((k&16) | (n&15)))*16 + (k&15)
// Fragment read for n-subtile t: 32 contiguous bytes at ((t*32+lane)*16).

// load 16 f32 (4x float4 / b128) and convert to bf16 vector (v_cvt_pk_bf16)
__device__ __forceinline__ v16bf cvt16f(const float* p) {
  v16bf r;
#pragma unroll
  for (int i = 0; i < 4; ++i) {
    float4 f = ((const float4*)p)[i];
    r[4*i+0] = (bf16_t)f.x; r[4*i+1] = (bf16_t)f.y;
    r[4*i+2] = (bf16_t)f.z; r[4*i+3] = (bf16_t)f.w;
  }
  return r;
}

#define LDA 48   // row-major LDS A tile stride (96B: keeps 16B chunks aligned)

// ---------------------------------------------------------------------------
// Kernel 1: QKV projection + elu+1 + layout-transforming epilogues.
// grid (ROWS/64, QKV_N/64), 128 threads (4 waves), 64x64 tile per block.
// ---------------------------------------------------------------------------
__global__ void __launch_bounds__(128)
qkv_kernel(const float* __restrict__ x, const float* __restrict__ Wqkv,
           const float* __restrict__ bqkv,
           bf16_t* __restrict__ qsw, bf16_t* __restrict__ kTb,
           bf16_t* __restrict__ vsw) {
  __shared__ bf16_t sA[64 * LDA];     // x tile, row-major
  __shared__ bf16_t sB[4 * 32 * 16];  // W tile, fragment-swizzled
  __shared__ bf16_t sD[64 * 64];      // output tile staging (q/v paths)
  const int tid = threadIdx.x, lane = tid & 31, wv = tid >> 5;
  const int m0 = blockIdx.x * 64;     // token-row tile
  const int n0 = blockIdx.y * 64;     // output-feature tile

  v8f acc[4];
  const v8f vz = {0.f,0.f,0.f,0.f,0.f,0.f,0.f,0.f};
#pragma unroll
  for (int i = 0; i < 4; ++i) acc[i] = vz;

  const int ml  = tid >> 1;           // 0..63
  const int kl0 = (tid & 1) * 16;     // 0 or 16
  const int sBoff = (((ml >> 4) * 32) + (kl0 | (ml & 15))) * 16;

  const float* xp = x    + (size_t)(m0 + ml) * DIM + kl0;
  const float* wp = Wqkv + (size_t)(n0 + ml) * DIM + kl0;

  // register double-buffer: tile kk=0 preloaded
  v16bf ax = cvt16f(xp);
  v16bf bx = cvt16f(wp);

  for (int kk = 0; kk < DIM; kk += 32) {
    __syncthreads();                  // prior LDS readers done
    *(v16bf*)&sA[ml * LDA + kl0] = ax;
    *(v16bf*)&sB[sBoff]          = bx;
    __syncthreads();
    if (kk + 32 < DIM) {              // overlap next-tile loads with WMMAs
      ax = cvt16f(xp + kk + 32);
      bx = cvt16f(wp + kk + 32);
    }
    if (kk + 64 < DIM) {              // prime caches one more tile ahead
      __builtin_prefetch(xp + kk + 64, 0, 1);
      __builtin_prefetch(wp + kk + 64, 0, 1);
    }
    v16bf bfm = *(const v16bf*)&sB[(wv * 32 + lane) * 16];
    v16bf af0 = frag_a(sA,  0, 0, LDA, lane);
    v16bf af1 = frag_a(sA, 16, 0, LDA, lane);
    v16bf af2 = frag_a(sA, 32, 0, LDA, lane);
    v16bf af3 = frag_a(sA, 48, 0, LDA, lane);
    acc[0] = wmma_bf16(af0, bfm, acc[0]);
    acc[1] = wmma_bf16(af1, bfm, acc[1]);
    acc[2] = wmma_bf16(af2, bfm, acc[2]);
    acc[3] = wmma_bf16(af3, bfm, acc[3]);
  }

  // ---- epilogue (s, h uniform per block: tiles never straddle 768-splits) --
  const int s  = n0 / DIM;            // 0=q 1=k 2=v (scalar branch)
  const int hh = (n0 % DIM) >> 6;     // head (scalar)
  const int b  = m0 >> 12;            // batch (scalar)
  const int tok0 = m0 & (SEQ - 1);    // scalar, multiple of 64
  const size_t hb = (size_t)(b * NHEAD + hh);
  const int d = wv * 16 + (lane & 15);             // feature within head
  const float bias = bqkv[n0 + d];
  const int roff = (lane < 16) ? 0 : 8;

  if (s == 1) {
    // k: elu+1, store transposed (D,N) row-major; 8 consecutive tokens/lane
#pragma unroll
    for (int mi = 0; mi < 4; ++mi) {
      v8bf pk;
#pragma unroll
      for (int r = 0; r < 8; ++r) pk[r] = (bf16_t)elu1(acc[mi][r] + bias);
      *(v8bf*)&kTb[(hb * HDIM + d) * SEQ + tok0 + mi * 16 + roff] = pk;
    }
  } else {
    // q / v: scatter into LDS in target swizzled order, then linear b128 out
    __syncthreads();
#pragma unroll
    for (int mi = 0; mi < 4; ++mi) {
#pragma unroll
      for (int r = 0; r < 8; ++r) {
        int tl = mi * 16 + r + roff;              // token within tile 0..63
        float val = acc[mi][r] + bias;
        int off;
        if (s == 0)   // q: K3-B layout, k-dim = d, n-dim = token
          off = (((d >> 5) * 4 + ((tl >> 4) & 3)) * 32 +
                 ((d & 16) | (tl & 15))) * 16 + (d & 15);
        else          // v: K2-B layout, k-dim = token, n-dim = d
          off = (((tl >> 5) * 4 + (d >> 4)) * 32 +
                 ((tl & 16) | (d & 15))) * 16 + (tl & 15);
        sD[off] = (s == 0) ? (bf16_t)elu1(val) : (bf16_t)val;
      }
    }
    __syncthreads();
    bf16_t* dst = (s == 0)
        ? qsw + hb * (size_t)SEQ * HDIM + (size_t)(tok0 >> 6) * 4096
        : vsw + hb * (size_t)SEQ * HDIM + (size_t)(tok0 >> 5) * 2048;
#pragma unroll
    for (int j = 0; j < 2; ++j) {
      int idx = tid * 32 + j * 16;
      *(v16bf*)&dst[idx] = *(const v16bf*)&sD[idx];
    }
  }
}

// ---------------------------------------------------------------------------
// Kernel 2: per (b,h): kvT_aug = [ (k^T v)^T ; k_sum ; zeros ]  (80x64 bf16)
// A = kT (LDS-staged, k_sum fused via ds_add_f32); B = v fragments direct
// from global (pre-swizzled by K1). grid (96), 128 threads.
// ---------------------------------------------------------------------------
__global__ void __launch_bounds__(128)
kv_kernel(const bf16_t* __restrict__ kTb, const bf16_t* __restrict__ vsw,
          bf16_t* __restrict__ kvTb) {
  __shared__ bf16_t sA[64 * LDA];
  __shared__ float  sk[HDIM];
  const int tid = threadIdx.x, lane = tid & 31, wv = tid >> 5;
  const int bh = blockIdx.x;
  const bf16_t* kT = kTb + (size_t)bh * HDIM * SEQ;
  const bf16_t* vb = vsw + (size_t)bh * SEQ * HDIM;

  if (tid < HDIM) sk[tid] = 0.0f;

  v8f acc[4];
  const v8f vz = {0.f,0.f,0.f,0.f,0.f,0.f,0.f,0.f};
#pragma unroll
  for (int i = 0; i < 4; ++i) acc[i] = vz;

  const int dl  = tid >> 1;           // 0..63 (row of kT)
  const int nl0 = (tid & 1) * 16;
  const bf16_t* kp = kT + (size_t)dl * SEQ + nl0;

  v16bf ax = *(const v16bf*)&kp[0];   // double-buffer registers

  for (int kk = 0; kk < SEQ; kk += 32) {
    __syncthreads();
    *(v16bf*)&sA[dl * LDA + nl0] = ax;
    {
      float p = 0.0f;
#pragma unroll
      for (int e = 0; e < 16; ++e) p += (float)ax[e];
      atomicAdd(&sk[dl], p);          // ds_add_f32: fused k_sum
    }
    __syncthreads();
    if (kk + 32 < SEQ) ax = *(const v16bf*)&kp[kk + 32];
    if (kk + 64 < SEQ) __builtin_prefetch(kp + kk + 64, 0, 1);
    // B fragment: one 32B coalesced global load per lane
    v16bf bfm = *(const v16bf*)&vb[(((size_t)(kk >> 5) * 4 + wv) * 32 + lane) * 16];
    v16bf af0 = frag_a(sA,  0, 0, LDA, lane);
    v16bf af1 = frag_a(sA, 16, 0, LDA, lane);
    v16bf af2 = frag_a(sA, 32, 0, LDA, lane);
    v16bf af3 = frag_a(sA, 48, 0, LDA, lane);
    acc[0] = wmma_bf16(af0, bfm, acc[0]);
    acc[1] = wmma_bf16(af1, bfm, acc[1]);
    acc[2] = wmma_bf16(af2, bfm, acc[2]);
    acc[3] = wmma_bf16(af3, bfm, acc[3]);
  }
  __syncthreads();

  // lane owns column dv = kv^T row dv; each acc[mi] gives 8 contiguous d's
  bf16_t* kvt = kvTb + (size_t)bh * KVROWS * HDIM;
  const int dv = wv * 16 + (lane & 15);
  const int roff = (lane < 16) ? 0 : 8;
#pragma unroll
  for (int mi = 0; mi < 4; ++mi) {
    v8bf pk;
#pragma unroll
    for (int r = 0; r < 8; ++r) pk[r] = (bf16_t)acc[mi][r];
    *(v8bf*)&kvt[(size_t)dv * HDIM + mi * 16 + roff] = pk;
  }
  if (tid < HDIM) kvt[64 * HDIM + tid] = (bf16_t)sk[tid];  // k_sum row
  for (int i = tid; i < 15 * HDIM; i += 128)                // zero pad rows
    kvt[65 * HDIM + i] = (bf16_t)0.0f;
}

// ---------------------------------------------------------------------------
// Kernel 3: out^T = kvT_aug(80x64) @ q^T(64x64-tile); row 64 of the product
// is the normalizer (q . k_sum) for free. No LDS. grid (SEQ/64, 96).
// ---------------------------------------------------------------------------
__global__ void __launch_bounds__(128)
out_kernel(const bf16_t* __restrict__ qsw, const bf16_t* __restrict__ kvTb,
           bf16_t* __restrict__ yb) {
  const int tid = threadIdx.x, lane = tid & 31, wv = tid >> 5;
  const int bh = blockIdx.y;
  const int t0 = blockIdx.x * 64;
  const int b = bh / NHEAD, h = bh % NHEAD;
  const bf16_t* A  = kvTb + (size_t)bh * KVROWS * HDIM;
  const bf16_t* Bq = qsw + (size_t)bh * SEQ * HDIM + (size_t)(t0 >> 6) * 4096;

  v8f acc[5];
  const v8f vz = {0.f,0.f,0.f,0.f,0.f,0.f,0.f,0.f};
#pragma unroll
  for (int i = 0; i < 5; ++i) acc[i] = vz;

#pragma unroll
  for (int ks = 0; ks < 2; ++ks) {
    // B fragment: 32B coalesced global load (pre-swizzled q)
    v16bf bfm = *(const v16bf*)&Bq[((ks * 4 + wv) * 32 + lane) * 16];
#pragma unroll
    for (int mi = 0; mi < 5; ++mi)
      acc[mi] = wmma_bf16(frag_a(A, mi * 16, ks * 32, HDIM, lane),
                          bfm, acc[mi]);
  }

  // normalizer: D row 64 -> acc[4][0] on lanes 0..15; broadcast to upper half
  float nrm = __shfl(acc[4][0], lane & 15) + 1e-6f;

  const int tok = t0 + wv * 16 + (lane & 15);
  bf16_t* yrow = yb + ((size_t)(b * SEQ + tok)) * DIM + h * HDIM;
  const int roff = (lane < 16) ? 0 : 8;
#pragma unroll
  for (int mi = 0; mi < 4; ++mi) {
    v8bf pk;
#pragma unroll
    for (int r = 0; r < 8; ++r) pk[r] = (bf16_t)(acc[mi][r] / nrm);
    *(v8bf*)&yrow[mi * 16 + roff] = pk;   // 8 consecutive features
  }
}

// ---------------------------------------------------------------------------
// Kernel 4: final = y @ Wproj^T (+bias) -> f32, LDS-transposed b128 stores.
// grid (ROWS/64, DIM/64), 128 threads.
// ---------------------------------------------------------------------------
#define LDD 68
__global__ void __launch_bounds__(128)
proj_kernel(const bf16_t* __restrict__ yb, const float* __restrict__ Wp,
            const float* __restrict__ bp, float* __restrict__ out) {
  __shared__ bf16_t sA[64 * LDA];
  __shared__ bf16_t sB[4 * 32 * 16];
  __shared__ float  sD[64 * LDD];
  const int tid = threadIdx.x, lane = tid & 31, wv = tid >> 5;
  const int m0 = blockIdx.x * 64;
  const int n0 = blockIdx.y * 64;

  v8f acc[4];
  const v8f vz = {0.f,0.f,0.f,0.f,0.f,0.f,0.f,0.f};
#pragma unroll
  for (int i = 0; i < 4; ++i) acc[i] = vz;

  const int ml  = tid >> 1;
  const int kl0 = (tid & 1) * 16;
  const int sBoff = (((ml >> 4) * 32) + (kl0 | (ml & 15))) * 16;

  const bf16_t* yp = yb + (size_t)(m0 + ml) * DIM + kl0;
  const float*  wp = Wp + (size_t)(n0 + ml) * DIM + kl0;

  v16bf ax = *(const v16bf*)&yp[0];
  v16bf bx = cvt16f(wp);

  for (int kk = 0; kk < DIM; kk += 32) {
    __syncthreads();
    *(v16bf*)&sA[ml * LDA + kl0] = ax;
    *(v16bf*)&sB[sBoff]          = bx;
    __syncthreads();
    if (kk + 32 < DIM) {
      ax = *(const v16bf*)&yp[kk + 32];
      bx = cvt16f(wp + kk + 32);
    }
    if (kk + 64 < DIM) {
      __builtin_prefetch(yp + kk + 64, 0, 1);
      __builtin_prefetch(wp + kk + 64, 0, 1);
    }
    v16bf bfm = *(const v16bf*)&sB[(wv * 32 + lane) * 16];
    v16bf af0 = frag_a(sA,  0, 0, LDA, lane);
    v16bf af1 = frag_a(sA, 16, 0, LDA, lane);
    v16bf af2 = frag_a(sA, 32, 0, LDA, lane);
    v16bf af3 = frag_a(sA, 48, 0, LDA, lane);
    acc[0] = wmma_bf16(af0, bfm, acc[0]);
    acc[1] = wmma_bf16(af1, bfm, acc[1]);
    acc[2] = wmma_bf16(af2, bfm, acc[2]);
    acc[3] = wmma_bf16(af3, bfm, acc[3]);
  }

  // stage f32 tile in LDS, write back coalesced float4 rows
  const int nloc = wv * 16 + (lane & 15);
  const float bias = bp[n0 + nloc];
  const int roff = (lane < 16) ? 0 : 8;
  __syncthreads();
#pragma unroll
  for (int mi = 0; mi < 4; ++mi)
#pragma unroll
    for (int r = 0; r < 8; ++r)
      sD[(mi * 16 + r + roff) * LDD + nloc] = acc[mi][r] + bias;
  __syncthreads();
#pragma unroll
  for (int j = 0; j < 8; ++j) {
    int idx = tid * 32 + j * 4;
    int row = idx >> 6, col = idx & 63;
    float4 t;
    t.x = sD[row * LDD + col];     t.y = sD[row * LDD + col + 1];
    t.z = sD[row * LDD + col + 2]; t.w = sD[row * LDD + col + 3];
    *(float4*)&out[(size_t)(m0 + row) * DIM + n0 + col] = t;
  }
}

// ---------------------------------------------------------------------------
extern "C" void kernel_launch(void* const* d_in, const int* in_sizes, int n_in,
                              void* d_out, int out_size, void* d_ws, size_t ws_size,
                              hipStream_t stream) {
  const float* x     = (const float*)d_in[0];
  const float* Wqkv  = (const float*)d_in[1];
  const float* bqkv  = (const float*)d_in[2];
  const float* Wproj = (const float*)d_in[3];
  const float* bproj = (const float*)d_in[4];
  float* out = (float*)d_out;

  char* ws = (char*)d_ws;
  size_t off = 0;
  bf16_t* qsw  = (bf16_t*)(ws + off); off += (size_t)NBH * SEQ * HDIM * sizeof(bf16_t);
  bf16_t* kTb  = (bf16_t*)(ws + off); off += (size_t)NBH * HDIM * SEQ * sizeof(bf16_t);
  bf16_t* vsw  = (bf16_t*)(ws + off); off += (size_t)NBH * SEQ * HDIM * sizeof(bf16_t);
  bf16_t* yb   = (bf16_t*)(ws + off); off += (size_t)ROWS * DIM * sizeof(bf16_t);
  bf16_t* kvTb = (bf16_t*)(ws + off); off += (size_t)NBH * KVROWS * HDIM * sizeof(bf16_t);
  (void)ws_size; (void)in_sizes; (void)n_in; (void)out_size;

  dim3 blk(128);
  qkv_kernel <<<dim3(ROWS / 64, QKV_N / 64), blk, 0, stream>>>(x, Wqkv, bqkv, qsw, kTb, vsw);
  kv_kernel  <<<dim3(NBH),                   blk, 0, stream>>>(kTb, vsw, kvTb);
  out_kernel <<<dim3(SEQ / 64, NBH),         blk, 0, stream>>>(qsw, kvTb, yb);
  proj_kernel<<<dim3(ROWS / 64, DIM / 64),   blk, 0, stream>>>(yb, Wproj, bproj, out);
}
